// MullenbachModel_249108103233
// MI455X (gfx1250) — matrix operand order repaired and checked by reading, not compiled, not run on hardware
//
#include <hip/hip_runtime.h>
#include <math.h>

typedef __attribute__((ext_vector_type(16))) __bf16 v16bf;
typedef __attribute__((ext_vector_type(8)))  __bf16 v8bf;
typedef __attribute__((ext_vector_type(8)))  float  v8f;

#define B_    4
#define T_    4096
#define D_    512
#define L_    8921

#define MTILE 64          // labels per workgroup
#define NT    64          // tokens per streaming step (4 waves: 2 token-halves x 32 labels)
#define NTILES (T_ / NT)  // 64

// round-to-nearest-even f32 -> bf16 bits
__device__ __forceinline__ unsigned bfbits(float x) {
  union { float f; unsigned u; } v; v.f = x;
  return (v.u + 0x7FFFu + ((v.u >> 16) & 1u)) >> 16;
}
__device__ __forceinline__ unsigned pk2(float lo, float hi) {
  return bfbits(lo) | (bfbits(hi) << 16);
}

// two 16B chunks at p and p+hioff (elems) -> one 16-elem bf16 fragment
__device__ __forceinline__ v16bf ldfrag(const unsigned short* p, int hioff) {
  v8bf lo = *(const v8bf*)(p);
  v8bf hi = *(const v8bf*)(p + hioff);
  return __builtin_shufflevector(lo, hi, 0,1,2,3,4,5,6,7,8,9,10,11,12,13,14,15);
}

__device__ __forceinline__ float rmax16(float x) {
  x = fmaxf(x, __shfl_xor(x, 1)); x = fmaxf(x, __shfl_xor(x, 2));
  x = fmaxf(x, __shfl_xor(x, 4)); x = fmaxf(x, __shfl_xor(x, 8));
  return x;
}
__device__ __forceinline__ float rsum16(float x) {
  x += __shfl_xor(x, 1); x += __shfl_xor(x, 2);
  x += __shfl_xor(x, 4); x += __shfl_xor(x, 8);
  return x;
}

#define WMMA_BF16(A, Bm, C) \
  __builtin_amdgcn_wmma_f32_16x16x32_bf16(false, (A), false, (Bm), (short)0, (C), false, false)

// ---------------- pre-pass: inputs f32 -> bf16 (dense [B][T][512]) ----------------
__global__ __launch_bounds__(256)
void cvt_inputs_bf16(const float* __restrict__ in, unsigned short* __restrict__ outw) {
  int i = blockIdx.x * 256 + threadIdx.x;            // one uint4 (8 elems) per thread
  const float4* p = (const float4*)in + (size_t)i * 2;
  float4 a = p[0], c = p[1];
  ((uint4*)outw)[i] = make_uint4(pk2(a.x, a.y), pk2(a.z, a.w),
                                 pk2(c.x, c.y), pk2(c.z, c.w));
}

// ---------------- fused CAML attention ----------------
// K-major LDS layouts (no padding; 16-lane x 16B-stride fragment loads are conflict-free):
//   A (U/F): chunk(q = K/8, r = label 0..63) at byte (q*64 + r)*16
//   B (inp): chunk(q = K/8, tok  0..63)      at byte (q*64 + tok)*16
template <bool ASYNC>
__global__ __launch_bounds__(128)
void caml_fused_wmma(const float* __restrict__ inp,          // [B,T,D] f32 (sync path)
                     const unsigned short* __restrict__ wsbf,// [B,T,D] bf16 (async path)
                     const float* __restrict__ Uw,           // [L,D]
                     const float* __restrict__ Fw,           // [L,D]
                     const float* __restrict__ bias,         // [L]
                     float* __restrict__ out)                // [B,L]
{
  __shared__ unsigned short Ulds[MTILE * D_];       // 64 KiB
  __shared__ unsigned short Flds[MTILE * D_];       // 64 KiB
  __shared__ unsigned short Ilds[2][NT * D_];       // 2 x 64 KiB  (total 256 KiB)

  const int tid  = threadIdx.x;
  const int lane = tid & 31;
  const int wave = tid >> 5;
  const int b    = blockIdx.y;
  const int l0   = blockIdx.x * MTILE;

  const unsigned ibase0 = (unsigned)(size_t)(void*)&Ilds[0][0]; // low 32 bits of flat = LDS offset
  const unsigned ibase1 = (unsigned)(size_t)(void*)&Ilds[1][0];
  const unsigned long long gplane =
      (unsigned long long)(size_t)wsbf + (unsigned long long)b * T_ * D_ * 2ull;

  // stage one 64-token tile into Ilds[buf]: 4096 16B chunks, 32 per thread,
  // globally contiguous 16B per lane; LDS side scatters into K-major layout
  auto stage = [&](int buf, int t0) {
    if constexpr (ASYNC) {
      unsigned goff = (unsigned)(t0 * (D_ * 2)) + (unsigned)tid * 16u;
      unsigned loff = (buf ? ibase1 : ibase0)
                    + (unsigned)(tid & 63) * 1024u      // q * 64 chunks * 16B
                    + (unsigned)(tid >> 6) * 16u;       // tok * 16B
      #pragma unroll
      for (int j = 0; j < 32; ++j) {
        asm volatile("global_load_async_to_lds_b128 %0, %1, %2"
                     :: "v"(loff), "v"(goff), "s"(gplane) : "memory");
        goff += 128u * 16u;   // 128 lanes x 16B
        loff += 32u;          // tok += 2 -> +2 chunks
      }
    } else {
      #pragma unroll 4
      for (int j = 0; j < 32; ++j) {
        int linear = tid + 128 * j;        // 0..4095
        int tok = linear >> 6;             // 0..63
        int q   = linear & 63;             // K/8 group
        const float4* src = (const float4*)(inp + ((size_t)b * T_ + t0 + tok) * D_ + q * 8);
        float4 a = src[0], c = src[1];
        *(uint4*)&Ilds[buf][(q * 64 + tok) * 8] =
            make_uint4(pk2(a.x, a.y), pk2(a.z, a.w), pk2(c.x, c.y), pk2(c.z, c.w));
      }
    }
  };

  // kick off tile 0 before weight staging so the DMA overlaps it
  stage(0, 0);

  // ---- stage U / F tiles: f32 -> bf16 (RNE) into K-major layout, zero rows past L ----
  for (int it2 = 0; it2 < 32; ++it2) {
    int idx = tid + 128 * it2;             // 4096 chunks per matrix
    int q   = idx >> 6;                    // 0..63
    int r   = idx & 63;                    // 0..63
    int lab = l0 + r;
    float4 u0 = make_float4(0.f,0.f,0.f,0.f), u1 = u0, f0 = u0, f1 = u0;
    if (lab < L_) {
      const float4* up = (const float4*)(Uw + (size_t)lab * D_ + q * 8);
      const float4* fp = (const float4*)(Fw + (size_t)lab * D_ + q * 8);
      u0 = up[0]; u1 = up[1]; f0 = fp[0]; f1 = fp[1];
    }
    *(uint4*)&Ulds[(q * 64 + r) * 8] = make_uint4(pk2(u0.x,u0.y), pk2(u0.z,u0.w), pk2(u1.x,u1.y), pk2(u1.z,u1.w));
    *(uint4*)&Flds[(q * 64 + r) * 8] = make_uint4(pk2(f0.x,f0.y), pk2(f0.z,f0.w), pk2(f1.x,f1.y), pk2(f1.z,f1.w));
  }

  // wave -> (label half, token half)
  const int g   = lane >> 4;               // C half: rows i + 8*g
  const int r16 = lane & 15;               // C column
  const int lw  = (wave & 1) * 32;         // label base of this wave within WG
  const int tw  = (wave >> 1) * 32;        // token base of this wave within step

  const unsigned short* Au0 = Ulds + (lw + r16) * 8;       // A rows lw..lw+15
  const unsigned short* Au1 = Au0 + 128;                   // A rows lw+16..lw+31
  const unsigned short* Af0 = Flds + (lw + r16) * 8;
  const unsigned short* Af1 = Af0 + 128;
  const unsigned short* IbP[2] = { &Ilds[0][(tw + r16) * 8], &Ilds[1][(tw + r16) * 8] };

  // online-softmax state: 16 slots = (mt 0..1) x (i 0..7), replicated over 16 lanes/half
  float run_m[16], run_s[16], run_n[16];
  #pragma unroll
  for (int i = 0; i < 16; ++i) { run_m[i] = -__builtin_inff(); run_s[i] = 0.f; run_n[i] = 0.f; }

  for (int it = 0; it < NTILES; ++it) {
    const int buf = it & 1;
    if constexpr (ASYNC) { asm volatile("s_wait_asynccnt 0x0" ::: "memory"); }
    __syncthreads();   // tile `it` resident; everyone done reading buf^1
    if (it + 1 < NTILES) stage(buf ^ 1, (it + 1) * NT);

    const unsigned short* ib = IbP[buf];
    v8f sc00 = {}, sc01 = {}, sc10 = {}, sc11 = {};
    v8f vc00 = {}, vc01 = {}, vc10 = {}, vc11 = {};

    #pragma unroll
    for (int kc = 0; kc < 16; ++kc) {
      const int qa = (4 * kc + g)     * 512;   // A: lo chunk, hi at +2 chunks (+1024 elems)
      const int qb = (4 * kc + g)     * 512;   // B: lo chunk, hi at +2 chunks (+1024 elems)
      v16bf b0 = ldfrag(ib + qb, 1024);        // tokens tw+0..15
      v16bf b1 = ldfrag(ib + 128 + qb, 1024);  // tokens tw+16..31
      v16bf u0 = ldfrag(Au0 + qa, 1024);
      v16bf u1 = ldfrag(Au1 + qa, 1024);
      v16bf f0 = ldfrag(Af0 + qa, 1024);
      v16bf f1 = ldfrag(Af1 + qa, 1024);
      sc00 = WMMA_BF16(u0, b0, sc00);  sc01 = WMMA_BF16(u0, b1, sc01);
      sc10 = WMMA_BF16(u1, b0, sc10);  sc11 = WMMA_BF16(u1, b1, sc11);
      vc00 = WMMA_BF16(f0, b0, vc00);  vc01 = WMMA_BF16(f0, b1, vc01);
      vc10 = WMMA_BF16(f1, b0, vc10);  vc11 = WMMA_BF16(f1, b1, vc11);
    }

    // ---- online softmax over this wave's 32 tokens (two 16-col tiles + lane reduce) ----
    #pragma unroll
    for (int mt = 0; mt < 2; ++mt) {
      #pragma unroll
      for (int i = 0; i < 8; ++i) {
        const int sl = mt * 8 + i;
        float s0 = mt ? sc10[i] : sc00[i];
        float s1 = mt ? sc11[i] : sc01[i];
        float w0 = mt ? vc10[i] : vc00[i];
        float w1 = mt ? vc11[i] : vc01[i];
        float m = rmax16(fmaxf(s0, s1));
        float newm = fmaxf(run_m[sl], m);
        float p0 = __expf(s0 - newm);
        float p1 = __expf(s1 - newm);
        float p  = rsum16(p0 + p1);
        float pv = rsum16(p0 * w0 + p1 * w1);
        float corr = __expf(run_m[sl] - newm);   // 0 on first tile (-inf)
        run_s[sl] = run_s[sl] * corr + p;
        run_n[sl] = run_n[sl] * corr + pv;
        run_m[sl] = newm;
      }
    }
  }

  // ---- merge the two token-halves per label via LDS scratch (reuse Ilds[0]) ----
  float* scr_m = (float*)&Ilds[0][0];      // [half][64]
  float* scr_s = scr_m + 128;
  float* scr_n = scr_s + 128;
  __syncthreads();                          // all compute reads of Ilds done
  if (r16 == 0) {
    const int half = wave >> 1;
    #pragma unroll
    for (int mt = 0; mt < 2; ++mt) {
      #pragma unroll
      for (int i = 0; i < 8; ++i) {
        const int sl = mt * 8 + i;
        const int ll = lw + mt * 16 + i + 8 * g;   // label-local 0..63
        scr_m[half * 64 + ll] = run_m[sl];
        scr_s[half * 64 + ll] = run_s[sl];
        scr_n[half * 64 + ll] = run_n[sl];
      }
    }
  }
  __syncthreads();
  if (tid < 64) {
    int lab = l0 + tid;
    float m0 = scr_m[tid], m1 = scr_m[64 + tid];
    float M  = fmaxf(m0, m1);
    float e0 = __expf(m0 - M), e1 = __expf(m1 - M);
    float s  = scr_s[tid] * e0 + scr_s[64 + tid] * e1;
    float n  = scr_n[tid] * e0 + scr_n[64 + tid] * e1;
    if (lab < L_) out[(size_t)b * L_ + lab] = n / s + bias[lab];
  }
}

extern "C" void kernel_launch(void* const* d_in, const int* in_sizes, int n_in,
                              void* d_out, int out_size, void* d_ws, size_t ws_size,
                              hipStream_t stream) {
  const float* inp  = (const float*)d_in[0];   // inputs [B,T,D]
  const float* Uw   = (const float*)d_in[1];   // U_weight [L,D]
  const float* Fw   = (const float*)d_in[2];   // final_weight [L,D]
  const float* bias = (const float*)d_in[3];   // final_bias [L]
  float* out = (float*)d_out;                  // [B,L]

  const size_t need = (size_t)B_ * T_ * D_ * 2;   // bf16 copy of inputs
  dim3 grid((L_ + MTILE - 1) / MTILE, B_);

  if (ws_size >= need) {
    unsigned short* wsbf = (unsigned short*)d_ws;
    cvt_inputs_bf16<<<dim3((B_ * T_ * D_ / 8) / 256), dim3(256), 0, stream>>>(inp, wsbf);
    caml_fused_wmma<true><<<grid, dim3(128), 0, stream>>>(inp, wsbf, Uw, Fw, bias, out);
  } else {
    caml_fused_wmma<false><<<grid, dim3(128), 0, stream>>>(inp, nullptr, Uw, Fw, bias, out);
  }
}
